// CRF_12060268167597
// MI455X (gfx1250) — compile-verified, matrix operand order checked by reading, not run on hardware
//
#include <hip/hip_runtime.h>
#include <stdint.h>

// ---------------------------------------------------------------------------
// CRF Viterbi decode for MI455X (gfx1250, wave32, WMMA)
//   Phase 0: pre-convert W -> padded bf16 [32][128] (once, removes per-block
//            f32->bf16 conversion ALU from the GEMM)
//   Phase 1: emissions e = X . W^T    (bf16 WMMA, f32 accum, HBM-bound,
//            non-temporal X loads / E stores)
//   Phase 2: per-word Viterbi scan    (1 wave32 per word, T in registers,
//            readlane broadcasts, backpointers in LDS, LDS backtrack)
// ---------------------------------------------------------------------------

typedef __attribute__((ext_vector_type(16))) __bf16 v16bf;
typedef __attribute__((ext_vector_type(8)))  __bf16 v8bf;
typedef __attribute__((ext_vector_type(8)))  float  v8f;
typedef __attribute__((ext_vector_type(4)))  float  v4f;   // native vector for NT loads

#define NUM_LABELS 26
#define SEQ_LEN    512
#define DIM        128
#define KPAD       32          // labels padded to 32 for coalesced scan reads
#define WORDS_PER_BLOCK 8

// ===========================================================================
// Kernel 0: W (f32 [26][128]) -> Wbf (bf16 [32][128], rows 26..31 zero)
// ===========================================================================
__global__ __launch_bounds__(256) void crf_prep_w(
    const float* __restrict__ W, __bf16* __restrict__ Wbf)
{
  for (int i = threadIdx.x; i < KPAD * DIM; i += 256) {
    const int row = i >> 7;
    const int col = i & (DIM - 1);
    const float v = (row < NUM_LABELS) ? W[row * DIM + col] : 0.0f;
    Wbf[i] = (__bf16)v;
  }
}

// ===========================================================================
// Kernel 1: emissions GEMM.  E[p, k] = sum_d X[p, d] * W[k, d]
//   p = flattened (b, l) row index, k padded to 32 (pad cols produce 0).
//   Block = 256 threads = 8 waves; each wave -> one 16-row x 32-col tile.
//   v_wmma_f32_16x16x32_bf16, 2 N-tiles x 4 K-steps = 8 WMMA per wave.
// ===========================================================================
__global__ __launch_bounds__(256) void crf_emissions(
    const float*  __restrict__ X,    // [B*L, 128]
    const __bf16* __restrict__ Wbf,  // [32, 128] bf16, pre-padded
    float* __restrict__ E)           // [B*L, 32]
{
  const int lane = threadIdx.x & 31;
  const int wave = threadIdx.x >> 5;
  const int half = lane >> 4;      // 0: lanes 0-15, 1: lanes 16-31
  const int l16  = lane & 15;
  const long rowBase = (long)blockIdx.x * 128 + (long)wave * 16;

  // ---- A fragments: 16-bit A 16x32 layout (lanes 0-15: K 0..7 & 16..23;
  //      lanes 16-31: K 8..15 & 24..31), 4 K-blocks cover D=128.
  v16bf afrag[4];
  {
    const float* xr = X + (rowBase + l16) * DIM;
    #pragma unroll
    for (int kb = 0; kb < 4; ++kb) {
      const int d0 = kb * 32 + half * 8;
      const v4f x0 = __builtin_nontemporal_load((const v4f*)(xr + d0));
      const v4f x1 = __builtin_nontemporal_load((const v4f*)(xr + d0 + 4));
      const v4f x2 = __builtin_nontemporal_load((const v4f*)(xr + d0 + 16));
      const v4f x3 = __builtin_nontemporal_load((const v4f*)(xr + d0 + 20));
      v16bf a;
      #pragma unroll
      for (int e = 0; e < 4; ++e) {
        a[e]      = (__bf16)x0[e];
        a[4 + e]  = (__bf16)x1[e];
        a[8 + e]  = (__bf16)x2[e];
        a[12 + e] = (__bf16)x3[e];
      }
      afrag[kb] = a;
    }
  }

  #pragma unroll
  for (int nt = 0; nt < 2; ++nt) {
    const int n = nt * 16 + l16;                       // output label column
    v8f c = {};
    #pragma unroll
    for (int kb = 0; kb < 4; ++kb) {
      const int d0 = kb * 32 + half * 8;
      // B fragment: two direct 16B bf16 loads (no conversion ALU)
      const __bf16* wr = Wbf + n * DIM + d0;
      const v8bf blo = *(const v8bf*)(wr);
      const v8bf bhi = *(const v8bf*)(wr + 16);
      v16bf b;
      #pragma unroll
      for (int e = 0; e < 8; ++e) { b[e] = blo[e]; b[8 + e] = bhi[e]; }
      c = __builtin_amdgcn_wmma_f32_16x16x32_bf16(
              /*neg_a=*/false, afrag[kb], /*neg_b=*/false, b,
              /*c_mod=*/(short)0, c, /*reuse_a=*/false, /*reuse_b=*/false);
    }
    // C/D layout: VGPR r -> (lanes 0-15: M=r, N=lane) (lanes 16-31: M=8+r)
    #pragma unroll
    for (int r = 0; r < 8; ++r) {
      const int m = r + half * 8;
      __builtin_nontemporal_store(c[r],
          E + (rowBase + m) * KPAD + nt * 16 + l16);
    }
  }
}

// ===========================================================================
// Kernel 2: Viterbi scan + backtrack.  One wave32 per word.
//   lane y holds l[y]; T column in 26 VGPRs; e rows prefetched 8-deep;
//   backpointers in LDS ([512][32] u8 per word); uniform backtrack.
// ===========================================================================
__device__ __forceinline__ float bcastf(float v, int l) {
  return __int_as_float(__builtin_amdgcn_readlane(__float_as_int(v), l));
}

__global__ __launch_bounds__(256) void crf_scan(
    const float* __restrict__ E,   // [B*L, 32]
    const float* __restrict__ T,   // [26, 26]
    int* __restrict__ out)         // [B, 512]
{
  extern __shared__ unsigned char smem[];
  const int lane = threadIdx.x & 31;
  const int wave = threadIdx.x >> 5;
  const long word = (long)blockIdx.x * WORDS_PER_BLOCK + wave;

  uint8_t* bp = smem + (size_t)wave * (SEQ_LEN * 32);   // [512][32] u8

  // T column registers: Tcol[j] = T[j][y] for this lane's label y
  float Tcol[NUM_LABELS];
  {
    const int y = (lane < NUM_LABELS) ? lane : (NUM_LABELS - 1);
    #pragma unroll
    for (int j = 0; j < NUM_LABELS; ++j) Tcol[j] = T[j * NUM_LABELS + y];
  }

  const float* erow = E + (size_t)word * SEQ_LEN * KPAD + lane;
  float lprev = 0.0f;

  for (int lc = 1; lc <= 511; lc += 8) {
    // prefetch 8 emission rows (coalesced: one 128B line per row, NT)
    float ebuf[8];
    #pragma unroll
    for (int u = 0; u < 8; ++u)
      ebuf[u] = __builtin_nontemporal_load(erow + (size_t)(lc - 1 + u) * KPAD);

    #pragma unroll
    for (int u = 0; u < 8; ++u) {
      const int l = lc + u;
      if (l <= 511) {
        const float a = ebuf[u] + lprev;        // lane j: e[l-1][j] + l[j]
        float best = -3.0e38f;
        int bestj = 0;
        #pragma unroll
        for (int j = 0; j < NUM_LABELS; ++j) {
          const float s = Tcol[j] + bcastf(a, j);
          if (s > best) { best = s; bestj = j; }   // first-max tie break
        }
        lprev = best;                              // l[l][y]
        bp[(size_t)l * 32 + lane] = (uint8_t)bestj;
      }
    }
  }

  // final: score_y = e[511][y] + l[511][y]; uniform argmax over lanes 0..25
  const float fin = __builtin_nontemporal_load(erow + (size_t)511 * KPAD) + lprev;
  float best = -3.0e38f;
  int ylast = 0;
  #pragma unroll
  for (int j = 0; j < NUM_LABELS; ++j) {
    const float s = bcastf(fin, j);
    if (s > best) { best = s; ylast = j; }
  }

  // backtrack: y[i-1] = bp[i][y[i]]  (uniform across the wave; LDS chain)
  int* wout = out + (size_t)word * SEQ_LEN;
  if (lane == 0) wout[511] = ylast;
  int y = ylast;
  for (int i = 511; i >= 1; --i) {
    y = (int)bp[(size_t)i * 32 + y];
    if (lane == 0) wout[i - 1] = y;
  }
}

// ===========================================================================
extern "C" void kernel_launch(void* const* d_in, const int* in_sizes, int n_in,
                              void* d_out, int out_size, void* d_ws, size_t ws_size,
                              hipStream_t stream) {
  const float* X = (const float*)d_in[0];   // [4096, 512, 128] f32
  const float* W = (const float*)d_in[1];   // [26, 128] f32
  const float* T = (const float*)d_in[2];   // [26, 26] f32
  int* out = (int*)d_out;                   // [4096, 512] int32 labels

  const int B  = 4096;
  const int BL = B * SEQ_LEN;               // 2,097,152 rows

  // workspace layout: E [B*L, 32] f32 (268 MB), then Wbf [32][128] bf16 (8 KB)
  float*  E   = (float*)d_ws;
  __bf16* Wbf = (__bf16*)((char*)d_ws + (size_t)BL * KPAD * sizeof(float));

  // Phase 0: one-time W -> bf16 padded table
  crf_prep_w<<<dim3(1), dim3(256), 0, stream>>>(W, Wbf);

  // Phase 1: emissions (8 WMMA / wave, 128 rows / block)
  crf_emissions<<<dim3(BL / 128), dim3(256), 0, stream>>>(X, Wbf, E);

  // Phase 2: scan (8 words/block, 128 KB LDS backpointers)
  const size_t smemBytes = (size_t)WORDS_PER_BLOCK * SEQ_LEN * 32;  // 131072
  (void)hipFuncSetAttribute((const void*)crf_scan,
                            hipFuncAttributeMaxDynamicSharedMemorySize,
                            (int)smemBytes);
  crf_scan<<<dim3(B / WORDS_PER_BLOCK), dim3(256), smemBytes, stream>>>(E, T, out);
}
